// CompressiveLayer_10368051052676
// MI455X (gfx1250) — compile-verified, matrix operand order checked by reading, not run on hardware
//
#include <hip/hip_runtime.h>

// ---------------------------------------------------------------------------
// Types
// ---------------------------------------------------------------------------
typedef __attribute__((ext_vector_type(4)))  unsigned int u32x4;
typedef __attribute__((ext_vector_type(8)))  float        v8f;
typedef __attribute__((ext_vector_type(16))) __bf16       v16bf;
typedef unsigned short u16;

union Frag { u32x4 q[2]; v16bf v; };

__device__ __forceinline__ u16 f2bf(float f) {
    union { float f; unsigned u; } x; x.f = f;
    unsigned r = x.u + 0x7FFFu + ((x.u >> 16) & 1u);   // round-to-nearest-even
    return (u16)(r >> 16);
}

// Async 16B global -> LDS copy (per lane), tracked by ASYNCcnt.
// vdst VGPR holds the wave-relative LDS byte offset (= low 32 bits of the
// generic pointer to __shared__ memory on gfx1250).
__device__ __forceinline__ void async_g2l_b128(const u16* gsrc, u16* lds_dst) {
    unsigned lds_off = (unsigned)(unsigned long long)(uintptr_t)lds_dst;
    unsigned long long ga = (unsigned long long)(uintptr_t)gsrc;
    asm volatile("global_load_async_to_lds_b128 %0, %1, off"
                 :: "v"(lds_off), "v"(ga)
                 : "memory");
}

__device__ __forceinline__ void wait_asynccnt0() {
    asm volatile("s_wait_asynccnt 0x0" ::: "memory");
}

// Problem constants
#define DMODEL 1024
#define NHEAD  16
#define HDIM   64
#define FFDIM  4096
#define SEQ    512
#define BATCH  4
#define MEMN   256
#define CMEMN  128
#define KTOT   896            // SEQ + MEMN + CMEMN
#define ROWS   (KTOT * BATCH) // 3584
#define ZBATCH (BATCH * NHEAD)// 64

// ---------------------------------------------------------------------------
// Generic batched bf16 GEMM: C[M,N] = A[M,Kd] * B, fp32 accumulate via WMMA.
// opB==0: B is row-major [Kd,N]; opB==1: B given transposed, row-major [N,Kd].
// Block tile 128x64, K-tile 64. 256 threads = 8 waves, wave tile 32x32.
// A tile (and B tile when opB==1) staged with GLOBAL_LOAD_ASYNC_TO_LDS_B128.
// ---------------------------------------------------------------------------
__global__ __launch_bounds__(256)
void gemm_bf16_kernel(const u16* __restrict__ A, int lda, long long sA,
                      const u16* __restrict__ B, int ldb, long long sB, int opB,
                      float*     __restrict__ C, int ldc, long long sC,
                      u16*       __restrict__ Cb, int ldcb, long long sCb,
                      const float* __restrict__ bias, int relu, int Kd)
{
    __shared__ __align__(16) u16 As[128][72];
    __shared__ __align__(16) u16 Bs[64][72];

    const int tid = threadIdx.x;
    const long long bz = blockIdx.z;
    A += bz * sA;
    B += bz * sB;
    if (C)  C  += bz * sC;
    if (Cb) Cb += bz * sCb;

    const int bm = blockIdx.y * 128;
    const int bn = blockIdx.x * 64;
    const int wv   = tid >> 5;
    const int lane = tid & 31;
    const int lrow = lane & 15;   // row/col within 16-tile
    const int kh   = lane >> 4;   // which K-half this lane owns
    const int wm = (wv & 3) * 32;
    const int wn = (wv >> 2) * 32;

    v8f acc[2][2];
    {
        v8f z8 = {0.f,0.f,0.f,0.f,0.f,0.f,0.f,0.f};
        acc[0][0] = z8; acc[0][1] = z8; acc[1][0] = z8; acc[1][1] = z8;
    }

    for (int kt = 0; kt < Kd; kt += 64) {
        // stage A tile 128x64 (1024 16B chunks, 4 per thread) via async copy
        #pragma unroll
        for (int i = 0; i < 4; ++i) {
            int q   = tid + i * 256;
            int row = q >> 3;
            int kc  = (q & 7) * 8;
            async_g2l_b128(A + (size_t)(bm + row) * lda + kt + kc, &As[row][kc]);
        }
        if (kt + 64 < Kd)  // prefetch next A tile -> global_prefetch
            __builtin_prefetch(A + (size_t)(bm + (tid >> 1)) * lda + kt + 64, 0, 1);

        if (opB) {
            // B[N,Kd]: contiguous 16B chunks, async copy into LDS
            #pragma unroll
            for (int i = 0; i < 2; ++i) {
                int q  = tid + i * 256;
                int n  = q >> 3;
                int kc = (q & 7) * 8;
                async_g2l_b128(B + (size_t)(bn + n) * ldb + kt + kc, &Bs[n][kc]);
            }
        } else {
            // B[Kd,N]: transpose into LDS as [n][k]
            #pragma unroll
            for (int i = 0; i < 16; ++i) {
                int lin = i * 256 + tid;
                int n = lin & 63;
                int k = lin >> 6;
                Bs[n][k] = B[(size_t)(kt + k) * ldb + bn + n];
            }
        }
        wait_asynccnt0();
        __syncthreads();

        #pragma unroll
        for (int ks = 0; ks < 64; ks += 32) {
            Frag fa[2], fb[2];
            #pragma unroll
            for (int mi = 0; mi < 2; ++mi) {
                int row = wm + mi * 16 + lrow;
                fa[mi].q[0] = *(const u32x4*)&As[row][ks + kh * 8];
                fa[mi].q[1] = *(const u32x4*)&As[row][ks + 16 + kh * 8];
            }
            #pragma unroll
            for (int ni = 0; ni < 2; ++ni) {
                int n = wn + ni * 16 + lrow;
                fb[ni].q[0] = *(const u32x4*)&Bs[n][ks + kh * 8];
                fb[ni].q[1] = *(const u32x4*)&Bs[n][ks + 16 + kh * 8];
            }
            #pragma unroll
            for (int mi = 0; mi < 2; ++mi)
                #pragma unroll
                for (int ni = 0; ni < 2; ++ni)
                    acc[mi][ni] = __builtin_amdgcn_wmma_f32_16x16x32_bf16(
                        false, fa[mi].v, false, fb[ni].v,
                        (short)0, acc[mi][ni], false, false);
        }
        __syncthreads();
    }

    // epilogue: C/D 16x16 layout -> VGPR r holds M=r (lanes 0-15) / M=r+8
    #pragma unroll
    for (int mi = 0; mi < 2; ++mi) {
        #pragma unroll
        for (int ni = 0; ni < 2; ++ni) {
            int gcol = bn + wn + ni * 16 + lrow;
            float bval = bias ? bias[gcol] : 0.f;
            #pragma unroll
            for (int r = 0; r < 8; ++r) {
                int grow = bm + wm + mi * 16 + kh * 8 + r;
                float v = acc[mi][ni][r] + bval;
                if (relu) v = fmaxf(v, 0.f);
                if (C)  C [(size_t)grow * ldc  + gcol] = v;
                if (Cb) Cb[(size_t)grow * ldcb + gcol] = f2bf(v);
            }
        }
    }
}

// ---------------------------------------------------------------------------
// LayerNorm over last dim (D=1024), one block per row, optional residual add.
// Writes fp32 and/or bf16 outputs.
// ---------------------------------------------------------------------------
__global__ __launch_bounds__(256)
void ln_kernel(const float* __restrict__ x, const float* __restrict__ res,
               const float* __restrict__ g, const float* __restrict__ b,
               float* __restrict__ y, u16* __restrict__ yb)
{
    __shared__ float r1[256];
    __shared__ float r2[256];
    const size_t row = blockIdx.x;
    const float* xr = x + row * DMODEL;
    const float* rr = res ? res + row * DMODEL : nullptr;
    const int tid = threadIdx.x;

    float s1 = 0.f, s2 = 0.f;
    for (int i = tid; i < DMODEL; i += 256) {
        float v = xr[i] + (rr ? rr[i] : 0.f);
        s1 += v; s2 += v * v;
    }
    r1[tid] = s1; r2[tid] = s2;
    __syncthreads();
    for (int s = 128; s > 0; s >>= 1) {
        if (tid < s) { r1[tid] += r1[tid + s]; r2[tid] += r2[tid + s]; }
        __syncthreads();
    }
    const float mean = r1[0] * (1.f / DMODEL);
    const float var  = r2[0] * (1.f / DMODEL) - mean * mean;
    const float rstd = rsqrtf(var + 1e-5f);

    for (int i = tid; i < DMODEL; i += 256) {
        float v = xr[i] + (rr ? rr[i] : 0.f);
        float o = (v - mean) * rstd * g[i] + b[i];
        if (y)  y [row * DMODEL + i] = o;
        if (yb) yb[row * DMODEL + i] = f2bf(o);
    }
}

// fp32 -> bf16 elementwise
__global__ __launch_bounds__(256)
void cvt_kernel(const float* __restrict__ src, u16* __restrict__ dst)
{
    size_t i = (size_t)blockIdx.x * 256 + threadIdx.x;
    dst[i] = f2bf(src[i]);
}

// Split qkv [3584,3072], add rel biases, pack to per-(b,h) [K,64] bf16 panels.
__global__ __launch_bounds__(256)
void pack_qkv_kernel(const float* __restrict__ qkv,
                     const float* __restrict__ rwb, const float* __restrict__ rrb,
                     u16* __restrict__ qA, u16* __restrict__ qB,
                     u16* __restrict__ kp, u16* __restrict__ vp)
{
    size_t t = (size_t)blockIdx.x * 256 + threadIdx.x;   // over ROWS*1024
    int dd = (int)(t % 1024);
    int h = dd >> 6, d = dd & 63;
    size_t r = t / 1024;
    int kk = (int)(r >> 2);
    int b  = (int)(r & 3);
    size_t base = r * 3072;
    float qv = qkv[base + dd];
    float kv = qkv[base + 1024 + dd];
    float vv = qkv[base + 2048 + dd];
    size_t o = ((size_t)(b * NHEAD + h) * KTOT + kk) * HDIM + d;
    qA[o] = f2bf(qv + rwb[h * HDIM + d]);
    qB[o] = f2bf(qv + rrb[h * HDIM + d]);
    kp[o] = f2bf(kv);
    vp[o] = f2bf(vv);
}

// r_head_k [K,1024] fp32 -> per-(b,h) [K,64] bf16, replicated across b.
__global__ __launch_bounds__(256)
void pack_rhk_kernel(const float* __restrict__ rhk, u16* __restrict__ out)
{
    size_t t = (size_t)blockIdx.x * 256 + threadIdx.x;   // over KTOT*1024
    int dd = (int)(t % 1024);
    int h = dd >> 6, d = dd & 63;
    int k = (int)(t / 1024);
    u16 v = f2bf(rhk[t]);
    #pragma unroll
    for (int b = 0; b < BATCH; ++b)
        out[((size_t)(b * NHEAD + h) * KTOT + k) * HDIM + d] = v;
}

// logits = (AC + rel_shift(BD)) * scale, in place on AC.
__global__ __launch_bounds__(256)
void combine_shift_kernel(float* __restrict__ ac, const float* __restrict__ bd)
{
    size_t t = (size_t)blockIdx.x * 256 + threadIdx.x;   // over Z*K*K
    int j = (int)(t % KTOT);
    size_t r = t / KTOT;
    int i = (int)(r % KTOT);
    size_t z = r / KTOT;
    const float* B = bd + z * (size_t)KTOT * KTOT;
    float s;
    if (j <= i)           s = B[(size_t)i * KTOT + (KTOT + j - i - 1)];
    else if (j == i + 1)  s = 0.f;
    else                  s = B[(size_t)(i + 1) * KTOT + (j - i - 2)];
    ac[t] = (ac[t] + s) * 0.125f;   // 1/sqrt(64)
}

// Row softmax over K=896 elems; writes bf16 probabilities.
__global__ __launch_bounds__(256)
void softmax_kernel(const float* __restrict__ logits, u16* __restrict__ p)
{
    __shared__ float red[256];
    const size_t row = blockIdx.x;
    const float* lr = logits + row * KTOT;
    u16* pr = p + row * KTOT;
    const int tid = threadIdx.x;

    float m = -3.4e38f;
    for (int i = tid; i < KTOT; i += 256) m = fmaxf(m, lr[i]);
    red[tid] = m; __syncthreads();
    for (int s = 128; s > 0; s >>= 1) {
        if (tid < s) red[tid] = fmaxf(red[tid], red[tid + s]);
        __syncthreads();
    }
    m = red[0]; __syncthreads();

    float sum = 0.f;
    for (int i = tid; i < KTOT; i += 256) sum += __expf(lr[i] - m);
    red[tid] = sum; __syncthreads();
    for (int s = 128; s > 0; s >>= 1) {
        if (tid < s) red[tid] += red[tid + s];
        __syncthreads();
    }
    const float inv = 1.f / red[0];
    for (int i = tid; i < KTOT; i += 256)
        pr[i] = f2bf(__expf(lr[i] - m) * inv);
}

// attn_vec [b][h][K][64] fp32 -> [k*B+b][h*64+d] bf16 for the O projection.
__global__ __launch_bounds__(256)
void avpack_kernel(const float* __restrict__ av, u16* __restrict__ out)
{
    size_t t = (size_t)blockIdx.x * 256 + threadIdx.x;   // over ROWS*1024
    int dd = (int)(t % 1024);
    int h = dd >> 6, d = dd & 63;
    size_t r = t / 1024;
    int i = (int)(r >> 2);
    int b = (int)(r & 3);
    out[t] = f2bf(av[((size_t)(b * NHEAD + h) * KTOT + i) * HDIM + d]);
}

// ---------------------------------------------------------------------------
// Host-side orchestration
// ---------------------------------------------------------------------------
static inline void* carve(char*& p, size_t bytes) {
    void* r = (void*)p;
    p += (bytes + 255) & ~(size_t)255;
    return r;
}

extern "C" void kernel_launch(void* const* d_in, const int* in_sizes, int n_in,
                              void* d_out, int out_size, void* d_ws, size_t ws_size,
                              hipStream_t stream) {
    (void)in_sizes; (void)n_in; (void)out_size; (void)ws_size;

    const float* input_ids = (const float*)d_in[0];   // [S,B,D]
    const float* pos_emb   = (const float*)d_in[1];   // [K,D]
    const float* mem       = (const float*)d_in[2];   // [M,B,D]
    const float* c_mem     = (const float*)d_in[3];   // [CM,B,D]
    const float* ln1_w     = (const float*)d_in[4];
    const float* ln1_b     = (const float*)d_in[5];
    const float* ln2_w     = (const float*)d_in[6];
    const float* ln2_b     = (const float*)d_in[7];
    const float* qkv_w     = (const float*)d_in[8];   // [D,3072]
    const float* r_w       = (const float*)d_in[9];   // [D,1024]
    const float* r_w_bias  = (const float*)d_in[10];  // [H,DH]
    const float* r_r_bias  = (const float*)d_in[11];  // [H,DH]
    const float* o_w       = (const float*)d_in[12];  // [1024,D]
    const float* ff1_w     = (const float*)d_in[13];  // [D,FF]
    const float* ff1_b     = (const float*)d_in[14];
    const float* ff2_w     = (const float*)d_in[15];  // [FF,D]
    const float* ff2_b     = (const float*)d_in[16];
    float* out             = (float*)d_out;           // [K,B,D]

    char* p = (char*)d_ws;
    float* w_f32    = (float*)carve(p, (size_t)ROWS * DMODEL * 4);
    u16*   w_bf     = (u16*)  carve(p, (size_t)ROWS * DMODEL * 2);
    u16*   wq_bf    = (u16*)  carve(p, (size_t)DMODEL * 3072 * 2);
    u16*   wr_bf    = (u16*)  carve(p, (size_t)DMODEL * DMODEL * 2);
    u16*   wo_bf    = (u16*)  carve(p, (size_t)DMODEL * DMODEL * 2);
    u16*   wf1_bf   = (u16*)  carve(p, (size_t)DMODEL * FFDIM * 2);
    u16*   wf2_bf   = (u16*)  carve(p, (size_t)FFDIM * DMODEL * 2);
    u16*   pos_bf   = (u16*)  carve(p, (size_t)KTOT * DMODEL * 2);
    float* qkv_f32  = (float*)carve(p, (size_t)ROWS * 3072 * 4);
    float* rhk_f32  = (float*)carve(p, (size_t)KTOT * DMODEL * 4);
    u16*   qA_bf    = (u16*)  carve(p, (size_t)ZBATCH * KTOT * HDIM * 2);
    u16*   qB_bf    = (u16*)  carve(p, (size_t)ZBATCH * KTOT * HDIM * 2);
    u16*   k_bf     = (u16*)  carve(p, (size_t)ZBATCH * KTOT * HDIM * 2);
    u16*   v_bf     = (u16*)  carve(p, (size_t)ZBATCH * KTOT * HDIM * 2);
    u16*   rhk_bf   = (u16*)  carve(p, (size_t)ZBATCH * KTOT * HDIM * 2);
    float* sc_ac    = (float*)carve(p, (size_t)ZBATCH * KTOT * KTOT * 4);
    float* sc_bd    = (float*)carve(p, (size_t)ZBATCH * KTOT * KTOT * 4);
    u16*   p_bf     = (u16*)  carve(p, (size_t)ZBATCH * KTOT * KTOT * 2);
    float* av_f32   = (float*)carve(p, (size_t)ZBATCH * KTOT * HDIM * 4);
    u16*   avp_bf   = (u16*)  carve(p, (size_t)ROWS * DMODEL * 2);
    float* aout_f32 = (float*)carve(p, (size_t)ROWS * DMODEL * 4);
    u16*   ares_bf  = (u16*)  carve(p, (size_t)ROWS * DMODEL * 2);
    u16*   hid_bf   = (u16*)  carve(p, (size_t)ROWS * FFDIM * 2);

    const dim3 blk(256);

    // 1) weight / pos-emb converts to bf16
    cvt_kernel<<<dim3((DMODEL*3072)/256), blk, 0, stream>>>(qkv_w, wq_bf);
    cvt_kernel<<<dim3((DMODEL*DMODEL)/256), blk, 0, stream>>>(r_w, wr_bf);
    cvt_kernel<<<dim3((DMODEL*DMODEL)/256), blk, 0, stream>>>(o_w, wo_bf);
    cvt_kernel<<<dim3((DMODEL*FFDIM)/256), blk, 0, stream>>>(ff1_w, wf1_bf);
    cvt_kernel<<<dim3((FFDIM*DMODEL)/256), blk, 0, stream>>>(ff2_w, wf2_bf);
    cvt_kernel<<<dim3((KTOT*DMODEL)/256), blk, 0, stream>>>(pos_emb, pos_bf);

    // 2) LN1 on input / mem / c_mem into w (rows are k*B+b)
    ln_kernel<<<dim3(SEQ*BATCH), blk, 0, stream>>>(input_ids, nullptr, ln1_w, ln1_b,
                                                   w_f32, w_bf);
    ln_kernel<<<dim3(MEMN*BATCH), blk, 0, stream>>>(mem, nullptr, ln1_w, ln1_b,
        w_f32 + (size_t)SEQ*BATCH*DMODEL, w_bf + (size_t)SEQ*BATCH*DMODEL);
    ln_kernel<<<dim3(CMEMN*BATCH), blk, 0, stream>>>(c_mem, nullptr, ln1_w, ln1_b,
        w_f32 + (size_t)(SEQ+MEMN)*BATCH*DMODEL, w_bf + (size_t)(SEQ+MEMN)*BATCH*DMODEL);

    // 3) QKV projection: [3584,1024] @ [1024,3072]
    gemm_bf16_kernel<<<dim3(3072/64, ROWS/128, 1), blk, 0, stream>>>(
        w_bf, DMODEL, 0, wq_bf, 3072, 0, 0,
        qkv_f32, 3072, 0, nullptr, 0, 0, nullptr, 0, DMODEL);

    // 4) r_head_k: [896,1024] @ [1024,1024]
    gemm_bf16_kernel<<<dim3(DMODEL/64, KTOT/128, 1), blk, 0, stream>>>(
        pos_bf, DMODEL, 0, wr_bf, DMODEL, 0, 0,
        rhk_f32, DMODEL, 0, nullptr, 0, 0, nullptr, 0, DMODEL);

    // 5) pack q(+biases)/k/v and r_head_k into per-(b,h) bf16 panels
    pack_qkv_kernel<<<dim3((ROWS*DMODEL)/256), blk, 0, stream>>>(
        qkv_f32, r_w_bias, r_r_bias, qA_bf, qB_bf, k_bf, v_bf);
    pack_rhk_kernel<<<dim3((KTOT*DMODEL)/256), blk, 0, stream>>>(rhk_f32, rhk_bf);

    // 6) AC = qA @ k^T  (batched over b,h; opB=1 since k is [K,64])
    const long long sQK = (long long)KTOT * HDIM;
    const long long sSC = (long long)KTOT * KTOT;
    gemm_bf16_kernel<<<dim3(KTOT/64, KTOT/128, ZBATCH), blk, 0, stream>>>(
        qA_bf, HDIM, sQK, k_bf, HDIM, sQK, 1,
        sc_ac, KTOT, sSC, nullptr, 0, 0, nullptr, 0, HDIM);

    // 7) BD = qB @ r_head_k^T
    gemm_bf16_kernel<<<dim3(KTOT/64, KTOT/128, ZBATCH), blk, 0, stream>>>(
        qB_bf, HDIM, sQK, rhk_bf, HDIM, sQK, 1,
        sc_bd, KTOT, sSC, nullptr, 0, 0, nullptr, 0, HDIM);

    // 8) logits = (AC + rel_shift(BD)) * scale; softmax -> bf16 P
    combine_shift_kernel<<<dim3((unsigned)((size_t)ZBATCH*KTOT*KTOT/256)), blk, 0, stream>>>(
        sc_ac, sc_bd);
    softmax_kernel<<<dim3(ZBATCH*KTOT), blk, 0, stream>>>(sc_ac, p_bf);

    // 9) attn_vec = P @ V  (batched; V row-major [K,64])
    gemm_bf16_kernel<<<dim3(HDIM/64, KTOT/128, ZBATCH), blk, 0, stream>>>(
        p_bf, KTOT, sSC, v_bf, HDIM, sQK, 0,
        av_f32, HDIM, sQK, nullptr, 0, 0, nullptr, 0, KTOT);

    // 10) repack attn_vec to [k*B+b, h*64+d] bf16, O projection
    avpack_kernel<<<dim3((ROWS*DMODEL)/256), blk, 0, stream>>>(av_f32, avp_bf);
    gemm_bf16_kernel<<<dim3(DMODEL/64, ROWS/128, 1), blk, 0, stream>>>(
        avp_bf, DMODEL, 0, wo_bf, DMODEL, 0, 0,
        aout_f32, DMODEL, 0, nullptr, 0, 0, nullptr, 0, DMODEL);

    // 11) post-LN residual: LN2(w + attn_out) -> bf16
    ln_kernel<<<dim3(ROWS), blk, 0, stream>>>(w_f32, aout_f32, ln2_w, ln2_b,
                                              nullptr, ares_bf);

    // 12) FF1 (+bias, ReLU) -> bf16 hidden; FF2 (+bias) -> fp32 output
    gemm_bf16_kernel<<<dim3(FFDIM/64, ROWS/128, 1), blk, 0, stream>>>(
        ares_bf, DMODEL, 0, wf1_bf, FFDIM, 0, 0,
        nullptr, 0, 0, hid_bf, FFDIM, 0, ff1_b, 1, DMODEL);
    gemm_bf16_kernel<<<dim3(DMODEL/64, ROWS/128, 1), blk, 0, stream>>>(
        hid_bf, FFDIM, 0, wf2_bf, DMODEL, 0, 0,
        out, DMODEL, 0, nullptr, 0, 0, ff2_b, 0, FFDIM);
}
